// AttentionContext_16054587752802
// MI455X (gfx1250) — matrix operand order repaired and checked by reading, not compiled, not run on hardware
//
#include <hip/hip_runtime.h>

// bf16 WMMA attention for gfx1250 (wave32, v_wmma_f32_16x16x32_bf16),
// with GLOBAL_LOAD_ASYNC_TO_LDS staging of K/V tiles in the attention kernel.
typedef __attribute__((ext_vector_type(16))) __bf16 v16bf;
typedef __attribute__((ext_vector_type(8)))  __bf16 v8bf;
typedef __attribute__((ext_vector_type(8)))  float  v8f;

#define SLAB_ELEMS (2048 * 64)   // one head-slab: [2048 rows x 64 cols]

// ---------- WMMA fragment helpers (contiguous k order in A and B: k = 16*(L>>4) + e) ----------

// A fragment 16x32 bf16 from row-major matrix (ld elems/row).
// lane L: row = r0 + (L&15); 16 contiguous elems at K = c0 + 16*(L>>4) + e.
__device__ __forceinline__ v16bf load_a_frag_bf(const __bf16* p, int ld, int r0, int c0, int lane) {
    int m = lane & 15, hh = lane >> 4;
    const __bf16* row = p + (size_t)(r0 + m) * ld + c0 + 16 * hh;
    v8bf lo = *(const v8bf*)(row);
    v8bf hi = *(const v8bf*)(row + 8);
    v16bf a;
#pragma unroll
    for (int e = 0; e < 8; ++e) { a[e] = lo[e]; a[8 + e] = hi[e]; }
    return a;
}

// Same A fragment but converting from f32 source on the fly.
__device__ __forceinline__ v16bf load_a_frag_f32(const float* p, int ld, int r0, int c0, int lane) {
    int m = lane & 15, hh = lane >> 4;
    const float* row = p + (size_t)(r0 + m) * ld + c0 + 16 * hh;
    v16bf a;
#pragma unroll
    for (int e = 0; e < 8; ++e) { a[e] = (__bf16)row[e]; a[8 + e] = (__bf16)row[8 + e]; }
    return a;
}

// B fragment 32x16 bf16: memory holds Bm[n][k] (row-major, ld elems/row) with
// B[k][n] = Bm[n][k]. lane L: n = n0+(L&15); 16 contiguous halfs at k = k0+16*(L>>4)+e.
__device__ __forceinline__ v16bf load_b_frag_bf(const __bf16* p, int ld, int n0, int k0, int lane) {
    int n = lane & 15, hh = lane >> 4;
    return *(const v16bf*)(p + (size_t)(n0 + n) * ld + k0 + 16 * hh);
}

__device__ __forceinline__ v8f wmma_bf16(v16bf a, v16bf b, v8f c) {
    return __builtin_amdgcn_wmma_f32_16x16x32_bf16(false, a, false, b, (short)0, c, false, false);
}

// Async 16-byte global->LDS copy (ASYNCcnt-tracked, CDNA5 ISA).
// lds_addr: wave-relative LDS byte address (low 32 bits of a generic LDS pointer).
__device__ __forceinline__ void async_copy16(unsigned lds_addr, const void* gptr) {
    asm volatile("global_load_async_to_lds_b128 %0, %1, off"
                 :: "v"(lds_addr), "v"((unsigned long long)(size_t)gptr)
                 : "memory");
}

// ---------- Kernel 1: convert all weights f32 -> bf16 (row-major kept) ----------
__global__ void __launch_bounds__(256) wconv_kernel(const float* wq, const float* wk, const float* wv,
                                                    const float* wo, __bf16* wqb, __bf16* wkb,
                                                    __bf16* wvb, __bf16* wob) {
    int i = blockIdx.x * 256 + threadIdx.x;
    if (i < 64 * 64) {
        wqb[i] = (__bf16)wq[i];
        wkb[i] = (__bf16)wk[i];
        wvb[i] = (__bf16)wv[i];
    }
    for (int j = i; j < 1024 * 1024; j += gridDim.x * 256) wob[j] = (__bf16)wo[j];
}

// ---------- Kernel 2: fused QKV projection per slab; Q pre-scaled by 1/8 ----------
// Q, K stored bf16 row-major [2048][64] per slab; V stored transposed [64][2048].
__global__ void __launch_bounds__(128) qkv_kernel(const float* x, const __bf16* wq, const __bf16* wk,
                                                  const __bf16* wv, const float* bq, const float* bk,
                                                  const float* bv, __bf16* Qb, __bf16* Kb, __bf16* Vtb) {
    int lane = threadIdx.x & 31, wave = threadIdx.x >> 5;
    int t = blockIdx.x * 4 + wave;          // 0..4095
    int slab = t >> 7, rt = t & 127;
    int row0 = rt * 16;
    int m16 = lane & 15, hh = lane >> 4;

    const float* xs = x + (size_t)slab * SLAB_ELEMS;
    __bf16* Qs = Qb + (size_t)slab * SLAB_ELEMS;
    __bf16* Ks = Kb + (size_t)slab * SLAB_ELEMS;
    __bf16* Vt = Vtb + (size_t)slab * SLAB_ELEMS;

    v16bf a0 = load_a_frag_f32(xs, 64, row0, 0, lane);
    v16bf a1 = load_a_frag_f32(xs, 64, row0, 32, lane);

#pragma unroll
    for (int nt = 0; nt < 4; ++nt) {
        int n0 = nt * 16;
        v8f cq = {}, ck = {}, cv = {};
        cq = wmma_bf16(a0, load_b_frag_bf(wq, 64, n0, 0, lane), cq);
        cq = wmma_bf16(a1, load_b_frag_bf(wq, 64, n0, 32, lane), cq);
        ck = wmma_bf16(a0, load_b_frag_bf(wk, 64, n0, 0, lane), ck);
        ck = wmma_bf16(a1, load_b_frag_bf(wk, 64, n0, 32, lane), ck);
        cv = wmma_bf16(a0, load_b_frag_bf(wv, 64, n0, 0, lane), cv);
        cv = wmma_bf16(a1, load_b_frag_bf(wv, 64, n0, 32, lane), cv);
        float bqv = bq[n0 + m16], bkv = bk[n0 + m16], bvv = bv[n0 + m16];
        v8bf pk;
#pragma unroll
        for (int r = 0; r < 8; ++r) {
            size_t ro = (size_t)(row0 + r + 8 * hh) * 64 + n0 + m16;
            Qs[ro] = (__bf16)((cq[r] + bqv) * 0.125f);  // softmax scale folded into Q
            Ks[ro] = (__bf16)(ck[r] + bkv);
            pk[r]  = (__bf16)(cv[r] + bvv);
        }
        // transposed V store: 8 consecutive rows -> one packed 16B store
        *(v8bf*)(Vt + (size_t)(n0 + m16) * 2048 + row0 + 8 * hh) = pk;
    }
}

// ---------- Kernel 3: flash attention; block = 4 waves = 64 query rows of one slab.
// K/V tiles staged once per block into LDS via async copies; all WMMA B-fragments
// are then fed from LDS (ds_load_b128), cutting VMEM traffic 4x vs per-wave loads.
__global__ void __launch_bounds__(128) attn_kernel(const __bf16* Qb, const __bf16* Kb,
                                                   const __bf16* Vtb, __bf16* ctxb) {
    __shared__ __bf16 kbuf[32 * 64];         // K tile  [key 0..31][d 0..63]   4KB
    __shared__ __bf16 vbuf[64 * 32];         // Vt tile [d 0..63][key 0..31]   4KB
    __shared__ __bf16 pbuf[4][16 * 32];      // per-wave P transpose buffers   4KB

    int lane = threadIdx.x & 31, wave = threadIdx.x >> 5;
    int tid  = threadIdx.x;
    int t = blockIdx.x * 4 + wave;           // 4 consecutive q-tiles, same slab
    int slab = t >> 7, qt = t & 127;
    int row0 = qt * 16;
    int m16 = lane & 15, hh = lane >> 4;

    const __bf16* Qs = Qb + (size_t)slab * SLAB_ELEMS;
    const __bf16* Ks = Kb + (size_t)slab * SLAB_ELEMS;
    const __bf16* Vt = Vtb + (size_t)slab * SLAB_ELEMS;
    __bf16* Cs = ctxb + (size_t)slab * SLAB_ELEMS;
    __bf16* pb = &pbuf[wave][0];

    // wave-relative LDS byte addresses (low 32 bits of generic LDS pointers)
    unsigned kbase = (unsigned)(size_t)&kbuf[0];
    unsigned vbase = (unsigned)(size_t)&vbuf[0];

    v16bf aq0 = load_a_frag_bf(Qs, 64, row0, 0, lane);
    v16bf aq1 = load_a_frag_bf(Qs, 64, row0, 32, lane);

    float mrow[8], lrow[8];
#pragma unroll
    for (int r = 0; r < 8; ++r) { mrow[r] = -1e30f; lrow[r] = 0.f; }
    v8f o0 = {}, o1 = {}, o2 = {}, o3 = {};

    // per-thread staging chunks: K tile 256x16B, Vt tile 256x16B, 2+2 per thread
    int c1 = tid + 128;
    int vd0 = tid >> 2, vp0 = (tid & 3) * 8;   // Vt chunk 0: row d, 8-elem part
    int vd1 = c1 >> 2,  vp1 = (c1 & 3) * 8;    // Vt chunk 1

    for (int kb = 0; kb < 64; ++kb) {        // 32 keys per step
        int n0 = kb * 32;

        // ---- async-stage K[n0..n0+31][0..63] and Vt[0..63][n0..n0+31] into LDS ----
        const __bf16* ksrc = Ks + (size_t)n0 * 64;           // 4KB contiguous
        async_copy16(kbase + (unsigned)tid * 16u, ksrc + tid * 8);
        async_copy16(kbase + (unsigned)c1 * 16u,  ksrc + c1 * 8);
        async_copy16(vbase + (unsigned)(vd0 * 32 + vp0) * 2u,
                     Vt + (size_t)vd0 * 2048 + n0 + vp0);
        async_copy16(vbase + (unsigned)(vd1 * 32 + vp1) * 2u,
                     Vt + (size_t)vd1 * 2048 + n0 + vp1);
        if (kb < 63) __builtin_prefetch(Ks + (size_t)(n0 + 32 + m16) * 64, 0, 1);

        asm volatile("s_wait_asynccnt 0x0" ::: "memory");    // my copies landed in LDS
        __syncthreads();                                     // everyone's copies landed

        // ---- scores: two 16x16 tiles, K-dim 64 split into 2 WMMAs each (from LDS) ----
        v8f s0 = {}, s1 = {};
        s0 = wmma_bf16(aq0, load_b_frag_bf(kbuf, 64, 0, 0, lane), s0);
        s0 = wmma_bf16(aq1, load_b_frag_bf(kbuf, 64, 0, 32, lane), s0);
        s1 = wmma_bf16(aq0, load_b_frag_bf(kbuf, 64, 16, 0, lane), s1);
        s1 = wmma_bf16(aq1, load_b_frag_bf(kbuf, 64, 16, 32, lane), s1);

        // ---- online softmax; row M = r + 8*hh lives across 16 lanes of a half-wave ----
#pragma unroll
        for (int r = 0; r < 8; ++r) {
            float v0 = s0[r], v1 = s1[r];
            float mx = fmaxf(v0, v1);
#pragma unroll
            for (int off = 8; off > 0; off >>= 1) mx = fmaxf(mx, __shfl_xor(mx, off));
            float mnew  = fmaxf(mrow[r], mx);
            float alpha = __expf(mrow[r] - mnew);
            float p0 = __expf(v0 - mnew), p1 = __expf(v1 - mnew);
            float rs = p0 + p1;
#pragma unroll
            for (int off = 8; off > 0; off >>= 1) rs += __shfl_xor(rs, off);
            lrow[r] = lrow[r] * alpha + rs;
            mrow[r] = mnew;
            o0[r] *= alpha; o1[r] *= alpha; o2[r] *= alpha; o3[r] *= alpha;
            pb[(r + 8 * hh) * 32 + m16]      = (__bf16)p0;   // C-layout -> row-major P
            pb[(r + 8 * hh) * 32 + 16 + m16] = (__bf16)p1;
        }
        asm volatile("s_wait_dscnt 0x0" ::: "memory");       // LDS RAW before transpose read

        // ---- ctx += P(16x32) x V(32x64), B-fragments from LDS Vt tile ----
        v16bf ap = load_a_frag_bf(pb, 32, 0, 0, lane);
        o0 = wmma_bf16(ap, load_b_frag_bf(vbuf, 32, 0,  0, lane), o0);
        o1 = wmma_bf16(ap, load_b_frag_bf(vbuf, 32, 16, 0, lane), o1);
        o2 = wmma_bf16(ap, load_b_frag_bf(vbuf, 32, 32, 0, lane), o2);
        o3 = wmma_bf16(ap, load_b_frag_bf(vbuf, 32, 48, 0, lane), o3);

        __syncthreads();                                     // done reading before restage
    }

#pragma unroll
    for (int r = 0; r < 8; ++r) {
        float inv = 1.0f / lrow[r];
        size_t ro = (size_t)(row0 + r + 8 * hh) * 64 + m16;
        Cs[ro]      = (__bf16)(o0[r] * inv);
        Cs[ro + 16] = (__bf16)(o1[r] * inv);
        Cs[ro + 32] = (__bf16)(o2[r] * inv);
        Cs[ro + 48] = (__bf16)(o3[r] * inv);
    }
}

// ---------- Kernel 4: output projection [4096,1024] x Wo^T + bo -> f32 ----------
__global__ void __launch_bounds__(128) proj_kernel(const __bf16* ctx, const __bf16* wo,
                                                   const float* bo, float* out) {
    int lane = threadIdx.x & 31, wave = threadIdx.x >> 5;
    int t = blockIdx.x * 4 + wave;           // 0..4095
    int rt = t >> 4, ct = t & 15;
    int row0 = rt * 16, col0 = ct * 64;
    int m16 = lane & 15, hh = lane >> 4;

    v8f c0 = {}, c1 = {}, c2 = {}, c3 = {};
    for (int k0 = 0; k0 < 1024; k0 += 32) {
        v16bf a = load_a_frag_bf(ctx, 1024, row0, k0, lane);
        c0 = wmma_bf16(a, load_b_frag_bf(wo, 1024, col0,      k0, lane), c0);
        c1 = wmma_bf16(a, load_b_frag_bf(wo, 1024, col0 + 16, k0, lane), c1);
        c2 = wmma_bf16(a, load_b_frag_bf(wo, 1024, col0 + 32, k0, lane), c2);
        c3 = wmma_bf16(a, load_b_frag_bf(wo, 1024, col0 + 48, k0, lane), c3);
    }
    float b0 = bo[col0 + m16], b1 = bo[col0 + 16 + m16];
    float b2 = bo[col0 + 32 + m16], b3 = bo[col0 + 48 + m16];
#pragma unroll
    for (int r = 0; r < 8; ++r) {
        size_t ro = (size_t)(row0 + r + 8 * hh) * 1024 + col0 + m16;
        out[ro]      = c0[r] + b0;
        out[ro + 16] = c1[r] + b1;
        out[ro + 32] = c2[r] + b2;
        out[ro + 48] = c3[r] + b3;
    }
}

// ---------- Host launcher ----------
extern "C" void kernel_launch(void* const* d_in, const int* in_sizes, int n_in,
                              void* d_out, int out_size, void* d_ws, size_t ws_size,
                              hipStream_t stream) {
    (void)in_sizes; (void)n_in; (void)out_size; (void)ws_size;
    const float* x  = (const float*)d_in[0];
    const float* wq = (const float*)d_in[1];
    const float* bq = (const float*)d_in[2];
    const float* wk = (const float*)d_in[3];
    const float* bk = (const float*)d_in[4];
    const float* wv = (const float*)d_in[5];
    const float* bv = (const float*)d_in[6];
    const float* wo = (const float*)d_in[7];
    const float* bo = (const float*)d_in[8];
    float* out = (float*)d_out;

    char* ws = (char*)d_ws;
    __bf16* wqb = (__bf16*)(ws + 0);              //  8 KB
    __bf16* wkb = (__bf16*)(ws + 8192);           //  8 KB
    __bf16* wvb = (__bf16*)(ws + 16384);          //  8 KB
    __bf16* wob = (__bf16*)(ws + 24576);          //  2 MB
    __bf16* Qb  = (__bf16*)(ws + 2121728);        //  8 MB
    __bf16* Kb  = (__bf16*)(ws + 10510336);       //  8 MB
    __bf16* Vtb = (__bf16*)(ws + 18898944);       //  8 MB (transposed V)
    __bf16* ctx = (__bf16*)(ws + 27287552);       //  8 MB

    hipLaunchKernelGGL(wconv_kernel, dim3(512), dim3(256), 0, stream,
                       wq, wk, wv, wo, wqb, wkb, wvb, wob);
    hipLaunchKernelGGL(qkv_kernel, dim3(1024), dim3(128), 0, stream,
                       x, wqb, wkb, wvb, bq, bk, bv, Qb, Kb, Vtb);
    hipLaunchKernelGGL(attn_kernel, dim3(1024), dim3(128), 0, stream,
                       Qb, Kb, Vtb, ctx);
    hipLaunchKernelGGL(proj_kernel, dim3(1024), dim3(128), 0, stream,
                       ctx, wob, bo, out);
}
